// AdaptiveSourceFieldEEG_34359738588
// MI455X (gfx1250) — matrix operand order repaired and checked by reading, not compiled, not run on hardware
//
#include <hip/hip_runtime.h>
#include <math.h>

// ---------------------------------------------------------------- constants
#define B_      16
#define C_      64
#define T_      4000
#define W_      250
#define STRIDE_ 125
#define NT_     31          // (4000-250)/125 + 1
#define D_      256
#define S_      64
#define BT_     (B_*NT_)    // 496
#define TOKS_   (BT_*C_)    // 31744
#define HTOKS_  (BT_*32)    // 15872
#define EPS_    1e-4f
#define SHRINK_ 0.1f

typedef __bf16 bf16_t;
typedef __attribute__((ext_vector_type(16))) __bf16    v16bf;
typedef __attribute__((ext_vector_type(8)))  float     v8f;
typedef __attribute__((ext_vector_type(4)))  unsigned  uint4v;
typedef __attribute__((ext_vector_type(8)))  unsigned  uint8v;

__device__ __forceinline__ bf16_t f2bf(float f) {
  unsigned u = __builtin_bit_cast(unsigned, f);
  unsigned r = u + 0x7FFFu + ((u >> 16) & 1u);   // RNE
  unsigned short h = (unsigned short)(r >> 16);
  return __builtin_bit_cast(bf16_t, h);
}
__device__ __forceinline__ float gelu_exact(float x) {
  return 0.5f * x * (1.0f + erff(x * 0.70710678118654752f));
}

// WMMA B-fragment packed layout: block = (k/32, n/16) of 32x16 bf16, 512 elems.
// lane = (n&15) | ((k>>4)&1)<<4 ; within-lane element offset = k&15.
__device__ __forceinline__ long long packB_idx(int k, int n, int N) {
  int lane = (n & 15) | (((k >> 4) & 1) << 4);
  return ((long long)((k >> 5) * (N >> 4) + (n >> 4)) << 9) + (lane << 4) + (k & 15);
}

__device__ __forceinline__ v16bf loadA_frag(const bf16_t* Arow, int k0) {
  uint4v alo = *(const uint4v*)(Arow + k0);
  uint4v ahi = *(const uint4v*)(Arow + k0 + 8);
  return __builtin_bit_cast(v16bf,
           __builtin_shufflevector(alo, ahi, 0, 1, 2, 3, 4, 5, 6, 7));
}

// ---------------------------------------------------------------- WMMA GEMM
// flags (epilogue only): 1 = GELU, 2 = residual add, 4 = pack bf16 out as
// batched-B fragments with K=64 rows per batch, N=256.
#define GF_GELU        1
#define GF_RESID       2
#define GF_OUTB_PACK64 4

// BMODE: 0 = B pre-packed fragment layout ; 1 = B row-major transposed (B[n][k])
// 32x32 output tile per wave: 2x2 subtiles, 4 accumulators, 4 WMMAs / K-step.
template<int BMODE>
__global__ __launch_bounds__(128) void gemm_bf16_t(
    const bf16_t* __restrict__ A, int lda, long long sA,
    const bf16_t* __restrict__ Bm, int ldb, long long sB,
    const float* __restrict__ bias, const float* __restrict__ Cin,
    float* __restrict__ outF, bf16_t* __restrict__ outB,
    int ldc, long long sC, int M, int N, int K, float alpha, int flags)
{
  const int lane = threadIdx.x & 31;
  const int wave = threadIdx.x >> 5;
  const int tilesN2 = N >> 5;                 // 32-wide output tiles
  const int tilesM2 = (M + 31) >> 5;          // 32-tall output tiles
  const int tile = blockIdx.x * 4 + wave;
  if (tile >= tilesM2 * tilesN2) return;      // wave-uniform
  const int tm = tile / tilesN2, tn = tile % tilesN2;
  const long long batch = blockIdx.y;
  const bf16_t* Ab = A + batch * sA;
  const bf16_t* Bb = Bm + batch * sB;

  int m0 = tm * 32 + (lane & 15);
  int m1 = m0 + 16;
  if (m0 >= M) m0 = M - 1;
  if (m1 >= M) m1 = M - 1;
  const int half = lane >> 4;
  const bf16_t* Arow0 = Ab + (long long)m0 * lda + (half << 4);
  const bf16_t* Arow1 = Ab + (long long)m1 * lda + (half << 4);

  v8f acc00 = {}, acc01 = {}, acc10 = {}, acc11 = {};
  for (int k0 = 0; k0 < K; k0 += 32) {
    v16bf a0 = loadA_frag(Arow0, k0);
    v16bf a1 = loadA_frag(Arow1, k0);
    v16bf b0, b1;
    if (BMODE == 1) {
      const bf16_t* Brow = Bb + (long long)(tn * 32 + (lane & 15)) * ldb
                              + k0 + (half << 4);
      b0 = __builtin_bit_cast(v16bf, *(const uint8v*)(Brow));
      b1 = __builtin_bit_cast(v16bf, *(const uint8v*)(Brow + (long long)16 * ldb));
      if (k0 + 32 < K) __builtin_prefetch(Brow + 32, 0, 1);
    } else {
      const bf16_t* blk = Bb + ((((long long)(k0 >> 5)) * (N >> 4) + (tn << 1)) << 9)
                             + (lane << 4);
      b0 = __builtin_bit_cast(v16bf, *(const uint8v*)(blk));
      b1 = __builtin_bit_cast(v16bf, *(const uint8v*)(blk + 512));
      if (k0 + 32 < K) __builtin_prefetch(blk + ((long long)(N >> 4) << 9), 0, 1);
    }
    acc00 = __builtin_amdgcn_wmma_f32_16x16x32_bf16(false, a0, false, b0,
                                                    (short)0, acc00, false, false);
    acc01 = __builtin_amdgcn_wmma_f32_16x16x32_bf16(false, a0, false, b1,
                                                    (short)0, acc01, false, false);
    acc10 = __builtin_amdgcn_wmma_f32_16x16x32_bf16(false, a1, false, b0,
                                                    (short)0, acc10, false, false);
    acc11 = __builtin_amdgcn_wmma_f32_16x16x32_bf16(false, a1, false, b1,
                                                    (short)0, acc11, false, false);
  }

#pragma unroll
  for (int msub = 0; msub < 2; ++msub) {
#pragma unroll
    for (int nsub = 0; nsub < 2; ++nsub) {
      v8f acc = msub ? (nsub ? acc11 : acc10) : (nsub ? acc01 : acc00);
      int nn = tn * 32 + nsub * 16 + (lane & 15);
#pragma unroll
      for (int r = 0; r < 8; ++r) {
        int mm = tm * 32 + msub * 16 + r + (half << 3);
        if (mm >= M) continue;
        float v = acc[r] * alpha;
        if (bias) v += bias[nn];
        if (flags & GF_GELU) v = gelu_exact(v);
        long long idx = batch * sC + (long long)mm * ldc + nn;
        if (flags & GF_RESID) v += Cin[idx];
        if (outF) outF[idx] = v;
        if (outB) {
          if (flags & GF_OUTB_PACK64) {
            long long pidx = ((long long)(mm >> 6) << 14) + packB_idx(mm & 63, nn, 256);
            outB[pidx] = f2bf(v);
          } else {
            outB[idx] = f2bf(v);
          }
        }
      }
    }
  }
}

// ---------------------------------------------------------------- weight prep
__global__ void conv_pad(const float* __restrict__ src, bf16_t* __restrict__ dst,
                         int Ksrc, int Kpad, int N) {
  long long i = (long long)blockIdx.x * 256 + threadIdx.x;
  if (i >= (long long)Kpad * N) return;
  int n = (int)(i % N); int k = (int)(i / N);
  dst[i] = f2bf(k < Ksrc ? src[(long long)k * N + n] : 0.f);
}
__global__ void conv_pad_pack(const float* __restrict__ src, bf16_t* __restrict__ dst,
                              int Ksrc, int Kpad, int N) {
  long long i = (long long)blockIdx.x * 256 + threadIdx.x;
  if (i >= (long long)Kpad * N) return;
  int n = (int)(i % N); int k = (int)(i / N);
  dst[packB_idx(k, n, N)] = f2bf(k < Ksrc ? src[(long long)k * N + n] : 0.f);
}
__global__ void conv_transpose_pack(const float* __restrict__ src, bf16_t* __restrict__ dst,
                                    int O, int I) {
  long long i = (long long)blockIdx.x * 256 + threadIdx.x;
  if (i >= (long long)O * I) return;
  int n = (int)(i % O); int k = (int)(i / O);
  dst[packB_idx(k, n, O)] = f2bf(src[(long long)n * I + k]);
}
__global__ void wcomb_pack(const float* __restrict__ raw, const float* __restrict__ dif,
                           bf16_t* __restrict__ dst) {
  int i = blockIdx.x * 256 + threadIdx.x;     // k*256 + d
  if (i >= 256 * 256) return;
  int d = i & 255, k = i >> 8;
  float v = 0.f;
  if (k < W_) {
    v = raw[k * 256 + d];
    if (k >= 1)      v += dif[(k - 1) * 256 + d];
    if (k <= W_ - 2) v -= dif[k * 256 + d];
  }
  dst[packB_idx(k, d, 256)] = f2bf(v);
}
__global__ void adjn_kernel(const float* __restrict__ adj, bf16_t* __restrict__ dst) {
  int r = blockIdx.x, i = threadIdx.x;
  __shared__ float red[64];
  float v = adj[r * 64 + i];
  red[i] = v; __syncthreads();
  for (int s = 32; s > 0; s >>= 1) { if (i < s) red[i] += red[i + s]; __syncthreads(); }
  dst[r * 64 + i] = f2bf(v / fmaxf(red[0], 1e-6f));
}
__global__ void bias_add2(const float* a, const float* b, float* o) {
  int i = threadIdx.x; o[i] = a[i] + b[i];
}

// ---------------------------------------------------------------- frames + stats
__global__ void frames_stats_kernel(const float* __restrict__ x,
                                    bf16_t* __restrict__ framesB,
                                    float* __restrict__ stats) {
  int idx = blockIdx.x;                      // b*NT*C + t*C + c
  int c = idx % C_; int t = (idx / C_) % NT_; int b = idx / (C_ * NT_);
  const float* xr = x + ((long long)(b * C_ + c)) * T_ + t * STRIDE_;
  __shared__ float red[256];
  int tid = threadIdx.x;
  float v = (tid < W_) ? xr[tid] : 0.f;
  red[tid] = v; __syncthreads();
  for (int s = 128; s > 0; s >>= 1) { if (tid < s) red[tid] += red[tid + s]; __syncthreads(); }
  float mean = red[0] / (float)W_;
  __syncthreads();
  float cv = (tid < W_) ? (v - mean) : 0.f;
  framesB[(long long)idx * 256 + tid] = f2bf(cv);
  red[tid] = fabsf(cv); __syncthreads();
  for (int s = 128; s > 0; s >>= 1) { if (tid < s) red[tid] += red[tid + s]; __syncthreads(); }
  float am = red[0] / (float)W_; __syncthreads();
  red[tid] = cv * cv; __syncthreads();
  for (int s = 128; s > 0; s >>= 1) { if (tid < s) red[tid] += red[tid + s]; __syncthreads(); }
  float rms = sqrtf(red[0] / (float)W_); __syncthreads();
  float dv = (tid < W_ - 1) ? fabsf(xr[tid + 1] - xr[tid]) : 0.f;
  red[tid] = dv; __syncthreads();
  for (int s = 128; s > 0; s >>= 1) { if (tid < s) red[tid] += red[tid + s]; __syncthreads(); }
  float dm = red[0] / (float)(W_ - 1); __syncthreads();
  red[tid] = fabsf(cv); __syncthreads();
  for (int s = 128; s > 0; s >>= 1) { if (tid < s) red[tid] = fmaxf(red[tid], red[tid + s]); __syncthreads(); }
  if (tid == 0) {
    float* st = stats + (long long)idx * 4;
    st[0] = am; st[1] = rms; st[2] = dm; st[3] = red[0];
  }
}

__global__ void stats_mlp1(const float* __restrict__ stats, const float* __restrict__ w,
                           const float* __restrict__ b, bf16_t* __restrict__ out) {
  long long i = (long long)blockIdx.x * 256 + threadIdx.x;
  if (i >= (long long)TOKS_ * 256) return;
  int d = (int)(i & 255); long long tok = i >> 8;
  const float* s = stats + tok * 4;
  float acc = b[d];
#pragma unroll
  for (int j = 0; j < 4; ++j) acc += s[j] * w[j * 256 + d];
  out[i] = f2bf(gelu_exact(acc));
}

// ---------------------------------------------------------------- covariance fix-up
__global__ void cov_fix(float* __restrict__ cov, const int* __restrict__ mask) {
  int bt = blockIdx.x; int b = bt / NT_;
  float* Cv = cov + (long long)bt * 4096;
  int i = threadIdx.x;                       // row
  __shared__ float red[64]; __shared__ float trs, mds;
  float vi = (float)mask[b * 64 + i];
  for (int j = 0; j < 64; ++j) Cv[i * 64 + j] *= vi * (float)mask[b * 64 + j];
  red[i] = Cv[i * 65]; __syncthreads();
  if (i == 0) { float s = 0; for (int j = 0; j < 64; ++j) s += red[j]; trs = fmaxf(s, EPS_); }
  __syncthreads();
  float inv = 1.0f / trs;
  for (int j = 0; j < 64; ++j) Cv[i * 64 + j] *= inv;
  __syncthreads();
  red[i] = Cv[i * 65]; __syncthreads();
  if (i == 0) { float s = 0; for (int j = 0; j < 64; ++j) s += red[j]; mds = s / 64.f; }
  __syncthreads();
  for (int j = 0; j < 64; ++j) {
    float v = (1.f - SHRINK_) * Cv[i * 64 + j];
    if (i == j) v += SHRINK_ * mds + EPS_;
    Cv[i * 64 + j] = v;
  }
}

// ---------------------------------------------------------------- Jacobi eigh + log-matrix feats
__global__ __launch_bounds__(64) void eigh_logm(const float* __restrict__ cov,
                                                bf16_t* __restrict__ feats) {
  __shared__ float A[64][65];
  __shared__ float Vv[64][65];
  __shared__ float diag0[64];
  __shared__ float loglam[64];
  __shared__ float cs[2];
  int bt = blockIdx.x; int i = threadIdx.x;
  const float* Cv = cov + (long long)bt * 4096;
  for (int j = 0; j < 64; ++j) { A[i][j] = Cv[i * 64 + j]; Vv[i][j] = (i == j) ? 1.f : 0.f; }
  diag0[i] = Cv[i * 65];
  __syncthreads();
  for (int sweep = 0; sweep < 6; ++sweep) {
    for (int p = 0; p < 63; ++p) {
      for (int q = p + 1; q < 64; ++q) {
        if (i == 0) {
          float apq = A[p][q]; float c = 1.f, s = 0.f;
          if (fabsf(apq) > 1e-12f) {
            float tau = (A[q][q] - A[p][p]) / (2.f * apq);
            float tt = (tau >= 0.f ? 1.f : -1.f) / (fabsf(tau) + sqrtf(1.f + tau * tau));
            c = rsqrtf(1.f + tt * tt); s = tt * c;
          }
          cs[0] = c; cs[1] = s;
        }
        __syncthreads();
        float c = cs[0], s = cs[1];
        float ap = A[p][i], aq = A[q][i];
        A[p][i] = c * ap - s * aq;
        A[q][i] = s * ap + c * aq;
        __syncthreads();
        ap = A[i][p]; aq = A[i][q];
        A[i][p] = c * ap - s * aq;
        A[i][q] = s * ap + c * aq;
        float vp = Vv[i][p], vq = Vv[i][q];
        Vv[i][p] = c * vp - s * vq;
        Vv[i][q] = s * vp + c * vq;
        __syncthreads();
      }
    }
  }
  loglam[i] = logf(fmaxf(A[i][i], EPS_));
  __syncthreads();
  bf16_t* fr = feats + (long long)bt * 64 * 96;
  for (int j = 0; j < 64; ++j) {
    float s = 0.f;
    for (int k = 0; k < 64; ++k) s += Vv[i][k] * loglam[k] * Vv[j][k];
    fr[i * 96 + j] = f2bf(s);
  }
  fr[i * 96 + 64] = f2bf(logf(fmaxf(diag0[i], EPS_)));
  for (int j = 65; j < 96; ++j) fr[i * 96 + j] = f2bf(0.f);
}

// ---------------------------------------------------------------- layernorm (row = 256)
__global__ void ln_kernel(const float* __restrict__ x, const float* __restrict__ g,
                          const float* __restrict__ bb,
                          float* __restrict__ outF, bf16_t* __restrict__ outB) {
  long long row = blockIdx.x; int d = threadIdx.x;
  float v = x[row * 256 + d];
  __shared__ float red[256];
  red[d] = v; __syncthreads();
  for (int s = 128; s > 0; s >>= 1) { if (d < s) red[d] += red[d + s]; __syncthreads(); }
  float mean = red[0] / 256.f; __syncthreads();
  float dv = v - mean;
  red[d] = dv * dv; __syncthreads();
  for (int s = 128; s > 0; s >>= 1) { if (d < s) red[d] += red[d + s]; __syncthreads(); }
  float y = dv * rsqrtf(red[0] / 256.f + 1e-5f) * g[d] + bb[d];
  if (outF) outF[row * 256 + d] = y;
  if (outB) outB[row * 256 + d] = f2bf(y);
}

// sensor = bf16(a+b) packed as batched B fragments (K=64 rows / batch, N=256)
__global__ void add2bf16_pack(const float* __restrict__ a, const float* __restrict__ b,
                              bf16_t* __restrict__ o, long long n) {
  long long i = (long long)blockIdx.x * 256 + threadIdx.x;
  if (i >= n) return;
  int d = (int)(i & 255); long long r = i >> 8;
  int c = (int)(r & 63); long long bt = r >> 6;
  o[(bt << 14) + packB_idx(c, d, 256)] = f2bf(a[i] + b[i]);
}

// ---------------------------------------------------------------- depthwise time conv (K=5, pad 2) + gelu
__global__ void dwconv_gelu(const float* __restrict__ h, const float* __restrict__ wdw,
                            const float* __restrict__ bdw, bf16_t* __restrict__ out) {
  long long i = (long long)blockIdx.x * 256 + threadIdx.x;
  if (i >= (long long)TOKS_ * 256) return;
  int d = (int)(i & 255);
  long long r = i >> 8;                      // b*NT*S + t*S + s
  int s = (int)(r % S_); long long r2 = r / S_;
  int t = (int)(r2 % NT_); int b = (int)(r2 / NT_);
  float acc = bdw[d];
#pragma unroll
  for (int j = 0; j < 5; ++j) {
    int tt = t + j - 2;
    if (tt < 0 || tt >= NT_) continue;
    acc += h[(((long long)(b * NT_ + tt) * S_ + s) << 8) + d] * wdw[d * 5 + j];
  }
  out[i] = f2bf(gelu_exact(acc));
}

// ---------------------------------------------------------------- hemi split / combine
__global__ void hemi_split(const float* __restrict__ h, bf16_t* __restrict__ com,
                           bf16_t* __restrict__ dif) {
  long long i = (long long)blockIdx.x * 256 + threadIdx.x;   // bt*32*256
  if (i >= (long long)HTOKS_ * 256) return;
  int d = (int)(i & 255); long long r = i >> 8;
  int sp = (int)(r % 32); long long bt = r / 32;
  long long base = ((bt * S_ + sp) << 8) + d;
  float L = h[base], R = h[base + (32 << 8)];
  com[i] = f2bf(0.5f * (L + R));
  dif[i] = f2bf(0.5f * (L - R));
}
__global__ void hemi_combine(const float* __restrict__ src, const float* __restrict__ cu,
                             const float* __restrict__ du, float* __restrict__ out) {
  long long i = (long long)blockIdx.x * 256 + threadIdx.x;   // bt*64*256
  if (i >= (long long)TOKS_ * 256) return;
  int d = (int)(i & 255); long long r = i >> 8;
  int s = (int)(r % S_); long long bt = r / S_;
  int sp = (s < 32) ? s : s - 32;
  long long hidx = ((bt * 32 + sp) << 8) + d;
  float u = cu[hidx], v = du[hidx];
  out[i] = src[i] + ((s < 32) ? (u + v) : (u - v));
}

// ---------------------------------------------------------------- pooling + head
__global__ void pool3(const float* __restrict__ src, const float* __restrict__ e1,
                      const float* __restrict__ e2, float* __restrict__ pooled) {
  int i = blockIdx.x * 256 + threadIdx.x;    // 3*16*256
  if (i >= 3 * 16 * 256) return;
  int d = i & 255; int b = (i >> 8) & 15; int which = i >> 12;
  const float* p = which == 0 ? src : (which == 1 ? e1 : e2);
  float s = 0.f;
  for (int t = 0; t < NT_; ++t)
    for (int ss = 0; ss < S_; ++ss)
      s += p[(((long long)(b * NT_ + t) * S_ + ss) << 8) + d];
  pooled[i] = s / (float)(NT_ * S_);
}
__global__ void head_kernel(const float* __restrict__ pooled, const float* __restrict__ rw,
                            const float* __restrict__ rb, const float* __restrict__ hw,
                            const float* __restrict__ hb, float* __restrict__ out) {
  int i = threadIdx.x; if (i >= 64) return;
  int cls = i & 3; int b = i >> 2;
  const float* ps = pooled + b * 256;
  const float* p1 = pooled + 4096 + b * 256;
  const float* p2 = pooled + 8192 + b * 256;
  float l0 = rb[0], l1 = rb[1];
  for (int d = 0; d < 256; ++d) { l0 += ps[d] * rw[d * 2]; l1 += ps[d] * rw[d * 2 + 1]; }
  float mx = fmaxf(l0, l1);
  float e0 = expf(l0 - mx), e1 = expf(l1 - mx);
  float w0 = e0 / (e0 + e1), w1 = e1 / (e0 + e1);
  float acc = hb[cls];
  for (int d = 0; d < 256; ++d) acc += (w0 * p1[d] + w1 * p2[d]) * hw[d * 4 + cls];
  out[b * 4 + cls] = acc;
}

// ================================================================ host
enum {
  IN_X = 0, IN_MASK, IN_CW1, IN_CB1, IN_CW2, IN_CB2, IN_CLN_G, IN_CLN_B,
  IN_TW_RAW, IN_TB_RAW, IN_TW_DIFF, IN_TB_DIFF, IN_TW_S1, IN_TB_S1, IN_TW_S2, IN_TB_S2,
  IN_TLN_G, IN_TLN_B, IN_MIX, IN_ADJ, IN_E1_LNG_G, IN_E1_LNG_B, IN_E1_GM_W, IN_E1_GM_B,
  IN_E1_LNT_G, IN_E1_LNT_B, IN_E1_DW, IN_E1_DWB, IN_E1_PW, IN_E1_PWB, IN_E1_LNF_G, IN_E1_LNF_B,
  IN_E1_F1, IN_E1_F1B, IN_E1_F2, IN_E1_F2B, IN_H_LN_G, IN_H_LN_B, IN_H_CW1, IN_H_CB1,
  IN_H_CW2, IN_H_CB2, IN_H_DW1, IN_H_DB1, IN_H_DW2, IN_H_DB2, IN_H_F1, IN_H_F1B,
  IN_H_F2, IN_H_F2B, IN_R_W, IN_R_B, IN_HD_W, IN_HD_B
};

static void gemm_go(hipStream_t st, int bmode,
                    const bf16_t* A, int lda, long long sA,
                    const bf16_t* Bm, int ldb, long long sB,
                    const float* bias, const float* Cin,
                    float* outF, bf16_t* outB, int ldc, long long sC,
                    int M, int N, int K, float alpha, int flags, int batches) {
  int tiles = ((M + 31) / 32) * (N / 32);
  dim3 grid((tiles + 3) / 4, batches);
  if (bmode == 1)
    gemm_bf16_t<1><<<grid, dim3(128), 0, st>>>(A, lda, sA, Bm, ldb, sB, bias, Cin,
                                               outF, outB, ldc, sC, M, N, K, alpha, flags);
  else
    gemm_bf16_t<0><<<grid, dim3(128), 0, st>>>(A, lda, sA, Bm, ldb, sB, bias, Cin,
                                               outF, outB, ldc, sC, M, N, K, alpha, flags);
}

extern "C" void kernel_launch(void* const* d_in, const int* in_sizes, int n_in,
                              void* d_out, int out_size, void* d_ws, size_t ws_size,
                              hipStream_t stream) {
  (void)in_sizes; (void)n_in; (void)out_size; (void)ws_size;
  const float* X    = (const float*)d_in[IN_X];
  const int*   MASK = (const int*)d_in[IN_MASK];
  auto F = [&](int i) { return (const float*)d_in[i]; };

  // -------- workspace slots
  char* ws = (char*)d_ws;
  size_t off = 0;
  auto alloc = [&](size_t bytes) -> char* {
    off = (off + 255) & ~(size_t)255;
    char* p = ws + off; off += bytes; return p;
  };
  const size_t SZF = (size_t)TOKS_ * 256 * 4;   // f32 token slot (== 31744x512 bf16)
  const size_t SZB = (size_t)TOKS_ * 256 * 2;   // bf16 token slot
  float*  F0 = (float*)alloc(SZF);
  float*  F1 = (float*)alloc(SZF);
  float*  F2 = (float*)alloc(SZF);
  float*  F3 = (float*)alloc(SZF);
  float*  F4 = (float*)alloc(SZF);
  bf16_t* Bs0 = (bf16_t*)alloc(SZB);
  bf16_t* Bs1 = (bf16_t*)alloc(SZB);
  bf16_t* Bs2 = (bf16_t*)alloc(SZB);
  float*  covW   = (float*)alloc((size_t)BT_ * 4096 * 4);
  bf16_t* featsW = (bf16_t*)alloc((size_t)BT_ * 64 * 96 * 2);
  float*  statsW = (float*)alloc((size_t)TOKS_ * 4 * 4);
  float*  pooled = (float*)alloc(3 * 16 * 256 * 4);
  // weight arena (bf16): B-side operands are fragment-packed
  bf16_t* wcombP = (bf16_t*)alloc(256 * 256 * 2);
  bf16_t* cw1P   = (bf16_t*)alloc(96 * 256 * 2);
  bf16_t* cw2P   = (bf16_t*)alloc(256 * 256 * 2);
  bf16_t* tws2P  = (bf16_t*)alloc(256 * 256 * 2);
  bf16_t* mixA   = (bf16_t*)alloc(64 * 64 * 2);      // A-side, row-major
  bf16_t* adjnA  = (bf16_t*)alloc(64 * 64 * 2);      // A-side, row-major
  bf16_t* gmwP   = (bf16_t*)alloc(256 * 256 * 2);
  bf16_t* pwTP   = (bf16_t*)alloc(256 * 256 * 2);
  bf16_t* f1P    = (bf16_t*)alloc(256 * 512 * 2);
  bf16_t* f2P    = (bf16_t*)alloc(512 * 256 * 2);
  bf16_t* hcw1P  = (bf16_t*)alloc(256 * 256 * 2);
  bf16_t* hcw2P  = (bf16_t*)alloc(256 * 256 * 2);
  bf16_t* hdw1P  = (bf16_t*)alloc(256 * 256 * 2);
  bf16_t* hdw2P  = (bf16_t*)alloc(256 * 256 * 2);
  bf16_t* hf1P   = (bf16_t*)alloc(256 * 512 * 2);
  bf16_t* hf2P   = (bf16_t*)alloc(512 * 256 * 2);
  float*  tbcomb = (float*)alloc(256 * 4);

  // -------- weight prep
  auto cvtP = [&](const float* s, bf16_t* d, int Ks, int Kp, int N) {
    long long n = (long long)Kp * N;
    conv_pad_pack<<<dim3((unsigned)((n + 255) / 256)), dim3(256), 0, stream>>>(s, d, Ks, Kp, N);
  };
  cvtP(F(IN_CW1),    cw1P, 65, 96, 256);
  cvtP(F(IN_CW2),    cw2P, 256, 256, 256);
  cvtP(F(IN_TW_S2),  tws2P, 256, 256, 256);
  cvtP(F(IN_E1_GM_W),gmwP, 256, 256, 256);
  cvtP(F(IN_E1_F1),  f1P, 256, 256, 512);
  cvtP(F(IN_E1_F2),  f2P, 512, 512, 256);
  cvtP(F(IN_H_CW1),  hcw1P, 256, 256, 256);
  cvtP(F(IN_H_CW2),  hcw2P, 256, 256, 256);
  cvtP(F(IN_H_DW1),  hdw1P, 256, 256, 256);
  cvtP(F(IN_H_DW2),  hdw2P, 256, 256, 256);
  cvtP(F(IN_H_F1),   hf1P, 256, 256, 512);
  cvtP(F(IN_H_F2),   hf2P, 512, 512, 256);
  conv_pad<<<dim3(16), dim3(256), 0, stream>>>(F(IN_MIX), mixA, 64, 64, 64);
  conv_transpose_pack<<<dim3(256), dim3(256), 0, stream>>>(F(IN_E1_PW), pwTP, 256, 256);
  wcomb_pack<<<dim3(256), dim3(256), 0, stream>>>(F(IN_TW_RAW), F(IN_TW_DIFF), wcombP);
  adjn_kernel<<<dim3(64), dim3(64), 0, stream>>>(F(IN_ADJ), adjnA);
  bias_add2<<<dim3(1), dim3(256), 0, stream>>>(F(IN_TB_RAW), F(IN_TB_DIFF), tbcomb);

  // -------- frames (bf16, K padded to 256) + transient stats
  frames_stats_kernel<<<dim3(TOKS_), dim3(256), 0, stream>>>(X, Bs0, statsW);
  stats_mlp1<<<dim3(TOKS_), dim3(256), 0, stream>>>(statsW, F(IN_TW_S1), F(IN_TB_S1), Bs1);

  // transient tok = frames @ Wcomb + (tb_raw+tb_diff)  -> F0
  gemm_go(stream, 0, Bs0, 256, 0, wcombP, 0, 0, tbcomb, nullptr,
          F0, nullptr, 256, 0, TOKS_, 256, 256, 1.f, 0, 1);
  // tok += s_hidden @ tw_s2 + tb_s2
  gemm_go(stream, 0, Bs1, 256, 0, tws2P, 0, 0, F(IN_TB_S2), F0,
          F0, nullptr, 256, 0, TOKS_, 256, 256, 1.f, GF_RESID, 1);

  // cov = frames @ frames^T / (W-1)  (batched, B row-major transposed)
  gemm_go(stream, 1, Bs0, 256, 64 * 256, Bs0, 256, 64 * 256, nullptr, nullptr,
          covW, nullptr, 64, 4096, 64, 64, 256, 1.f / (float)(W_ - 1), 0, BT_);
  cov_fix<<<dim3(BT_), dim3(64), 0, stream>>>(covW, MASK);
  eigh_logm<<<dim3(BT_), dim3(64), 0, stream>>>(covW, featsW);

  // cov MLP
  gemm_go(stream, 0, featsW, 96, 0, cw1P, 0, 0, F(IN_CB1), nullptr,
          nullptr, Bs1, 256, 0, TOKS_, 256, 96, 1.f, GF_GELU, 1);
  gemm_go(stream, 0, Bs1, 256, 0, cw2P, 0, 0, F(IN_CB2), nullptr,
          F1, nullptr, 256, 0, TOKS_, 256, 256, 1.f, 0, 1);

  // sensor tokens = LN(tok) + LN(covh) -> packed bf16 (Bs2)
  ln_kernel<<<dim3(TOKS_), dim3(256), 0, stream>>>(F0, F(IN_TLN_G), F(IN_TLN_B), F0, nullptr);
  ln_kernel<<<dim3(TOKS_), dim3(256), 0, stream>>>(F1, F(IN_CLN_G), F(IN_CLN_B), F1, nullptr);
  add2bf16_pack<<<dim3(TOKS_), dim3(256), 0, stream>>>(F0, F1, Bs2, (long long)TOKS_ * 256);

  // src = mix (64x64, shared A) x sensor (packed batched B) -> F0
  gemm_go(stream, 0, mixA, 64, 0, Bs2, 0, 64 * 256, nullptr, nullptr,
          F0, nullptr, 256, 64 * 256, 64, 256, 64, 1.f, 0, BT_);

  // ================= event expert =================
  ln_kernel<<<dim3(TOKS_), dim3(256), 0, stream>>>(F0, F(IN_E1_LNG_G), F(IN_E1_LNG_B),
                                                   nullptr, Bs0);
  gemm_go(stream, 0, Bs0, 256, 0, gmwP, 0, 0, F(IN_E1_GM_B), nullptr,
          nullptr, Bs1, 256, 0, TOKS_, 256, 256, 1.f, GF_OUTB_PACK64, 1);   // gm packed
  gemm_go(stream, 0, adjnA, 64, 0, Bs1, 0, 64 * 256, nullptr, F0,
          F1, nullptr, 256, 64 * 256, 64, 256, 64, 1.f, GF_RESID, BT_);     // x1
  ln_kernel<<<dim3(TOKS_), dim3(256), 0, stream>>>(F1, F(IN_E1_LNT_G), F(IN_E1_LNT_B),
                                                   F2, nullptr);
  dwconv_gelu<<<dim3(TOKS_), dim3(256), 0, stream>>>(F2, F(IN_E1_DW), F(IN_E1_DWB), Bs2);
  gemm_go(stream, 0, Bs2, 256, 0, pwTP, 0, 0, F(IN_E1_PWB), F1,
          F2, nullptr, 256, 0, TOKS_, 256, 256, 1.f, GF_RESID, 1);          // x2
  ln_kernel<<<dim3(TOKS_), dim3(256), 0, stream>>>(F2, F(IN_E1_LNF_G), F(IN_E1_LNF_B),
                                                   nullptr, Bs0);
  bf16_t* fh = (bf16_t*)F1;
  gemm_go(stream, 0, Bs0, 256, 0, f1P, 0, 0, F(IN_E1_F1B), nullptr,
          nullptr, fh, 512, 0, TOKS_, 512, 256, 1.f, GF_GELU, 1);
  gemm_go(stream, 0, fh, 512, 0, f2P, 0, 0, F(IN_E1_F2B), F2,
          F3, nullptr, 256, 0, TOKS_, 256, 512, 1.f, GF_RESID, 1);          // e1out

  // ================= hemi expert =================
  ln_kernel<<<dim3(TOKS_), dim3(256), 0, stream>>>(F0, F(IN_H_LN_G), F(IN_H_LN_B),
                                                   F2, nullptr);
  hemi_split<<<dim3(HTOKS_), dim3(256), 0, stream>>>(F2, Bs0, Bs1);
  gemm_go(stream, 0, Bs0, 256, 0, hcw1P, 0, 0, F(IN_H_CB1), nullptr,
          nullptr, Bs2, 256, 0, HTOKS_, 256, 256, 1.f, GF_GELU, 1);
  float* cu = F1;
  gemm_go(stream, 0, Bs2, 256, 0, hcw2P, 0, 0, F(IN_H_CB2), nullptr,
          cu, nullptr, 256, 0, HTOKS_, 256, 256, 1.f, 0, 1);
  gemm_go(stream, 0, Bs1, 256, 0, hdw1P, 0, 0, F(IN_H_DB1), nullptr,
          nullptr, Bs2, 256, 0, HTOKS_, 256, 256, 1.f, GF_GELU, 1);
  float* du = F1 + (long long)HTOKS_ * 256;
  gemm_go(stream, 0, Bs2, 256, 0, hdw2P, 0, 0, F(IN_H_DB2), nullptr,
          du, nullptr, 256, 0, HTOKS_, 256, 256, 1.f, 0, 1);
  hemi_combine<<<dim3(TOKS_), dim3(256), 0, stream>>>(F0, cu, du, F2);
  ln_kernel<<<dim3(TOKS_), dim3(256), 0, stream>>>(F2, F(IN_H_LN_G), F(IN_H_LN_B),
                                                   nullptr, Bs0);
  bf16_t* fh2 = (bf16_t*)F1;
  gemm_go(stream, 0, Bs0, 256, 0, hf1P, 0, 0, F(IN_H_F1B), nullptr,
          nullptr, fh2, 512, 0, TOKS_, 512, 256, 1.f, GF_GELU, 1);
  gemm_go(stream, 0, fh2, 512, 0, hf2P, 0, 0, F(IN_H_F2B), F2,
          F4, nullptr, 256, 0, TOKS_, 256, 512, 1.f, GF_RESID, 1);          // e2out

  // ================= router + head =================
  pool3<<<dim3((3 * 16 * 256 + 255) / 256), dim3(256), 0, stream>>>(F0, F3, F4, pooled);
  head_kernel<<<dim3(1), dim3(64), 0, stream>>>(pooled, F(IN_R_W), F(IN_R_B),
                                                F(IN_HD_W), F(IN_HD_B), (float*)d_out);
}
